// SelfAttention_7481833030271
// MI455X (gfx1250) — compile-verified, matrix operand order checked by reading, not run on hardware
//
#include <hip/hip_runtime.h>

typedef _Float16 hv8  __attribute__((ext_vector_type(8)));
typedef _Float16 hv16 __attribute__((ext_vector_type(16)));
typedef float    fv8  __attribute__((ext_vector_type(8)));

#define NB    64
#define TOTB  8192
#define DD    64
#define DKK   32
#define CC    16
#define NHH   8
#define LL    512          // TOT / C
#define KDIM  1024         // C * D
#define NCOL  256          // NH * DK

// Pack W_q (NH,C,D,DK) fp32 -> f16, layout Wt[col][k] with col = h*32+t, k = c*64+d.
// Coalesced writes; scattered reads hit L2 (W_q is only 1 MiB).
__global__ __launch_bounds__(256) void wq_pack_f16(const float* __restrict__ Wq,
                                                   _Float16* __restrict__ Wt) {
    int o = blockIdx.x * 256 + threadIdx.x;
    if (o >= NHH * DKK * KDIM) return;
    int col = o >> 10;           // h*32 + t
    int k   = o & 1023;          // c*64 + d
    int h = col >> 5, t = col & 31;
    int c = k >> 6,  d = k & 63;
    Wt[o] = (_Float16)Wq[(((h * CC + c) * DD + d) * DKK) + t];
}

__global__ __launch_bounds__(512) void attn_fused(const float* __restrict__ x,
                                                  const _Float16* __restrict__ Wt,
                                                  const float* __restrict__ Wk,
                                                  float* __restrict__ out) {
    __shared__ __align__(16) _Float16 smX[16 * KDIM];   // 32 KB: X tile, f16
    __shared__ float smQ[16 * NCOL];                    // 16 KB: Q tile, fp32
    __shared__ float smA[16 * 64];                      //  4 KB: per-wave A scratch
    __shared__ float smS[16 * 16];                      //  1 KB: per-wave score scratch

    const int tid   = threadIdx.x;
    const int wave  = tid >> 5;
    const int lane  = tid & 31;
    const int n     = blockIdx.x >> 5;
    const int ltile = blockIdx.x & 31;

    // ---- Load 16x1024 x-tile: contiguous 64 KB, fp32 -> f16 LDS ----
    const float* xsrc = x + ((size_t)n * TOTB + (size_t)ltile * 256) * DD;
    for (int i = tid; i < 16 * KDIM / 4; i += 512) {
        float4 v = *(const float4*)(xsrc + (size_t)i * 4);
        int o = i * 4;
        smX[o + 0] = (_Float16)v.x;
        smX[o + 1] = (_Float16)v.y;
        smX[o + 2] = (_Float16)v.z;
        smX[o + 3] = (_Float16)v.w;
    }
    __syncthreads();

    // ---- GEMM1: Q[16 x 256] = X[16 x 1024] * Wt^T, one 16-col tile per wave ----
    // 16-bit A/B fragment K-offsets per ISA layout:
    //   lanes 0..15:  K = {kb+0..7, kb+16..23}; lanes 16..31: K = {kb+8..15, kb+24..31}
    fv8 acc = {};
    const int mn   = lane & 15;
    const int khi  = (lane >> 4) ? 8 : 0;
    const _Float16* Ap = smX + (size_t)mn * KDIM;                 // A row (M = mn)
    const _Float16* Bp = Wt + (size_t)(wave * 16 + mn) * KDIM;    // B col (N = mn)
#pragma unroll 4
    for (int kk = 0; kk < 32; ++kk) {
        const int kb = kk * 32 + khi;
        hv8 a0 = *(const hv8*)(Ap + kb);
        hv8 a1 = *(const hv8*)(Ap + kb + 16);
        hv8 b0 = *(const hv8*)(Bp + kb);
        hv8 b1 = *(const hv8*)(Bp + kb + 16);
        hv16 A = __builtin_shufflevector(a0, a1, 0,1,2,3,4,5,6,7,8,9,10,11,12,13,14,15);
        hv16 B = __builtin_shufflevector(b0, b1, 0,1,2,3,4,5,6,7,8,9,10,11,12,13,14,15);
        acc = __builtin_amdgcn_wmma_f32_16x16x32_f16(false, A, false, B,
                                                     (short)0, acc, false, false);
    }
    {   // C/D layout: VGPR r -> M = r (+8 for lanes 16..31), N = lane&15
        const int mbase = (lane >> 4) ? 8 : 0;
#pragma unroll
        for (int r = 0; r < 8; ++r)
            smQ[(mbase + r) * NCOL + wave * 16 + mn] = acc[r];
    }
    __syncthreads();

    // ---- Stage 2: per (h,l) pair -> A = Wk[h]*Q, QK = X*A, softmax, attn ----
    float* myA = smA + wave * 64;
    float* myS = smS + wave * 16;
    const float inv_sqrt_dk = 0.17677669529663687f;   // 1/sqrt(32)

    for (int p = 0; p < 8; ++p) {
        const int pr = wave * 8 + p;                  // 128 pairs / 16 waves
        const int h  = pr >> 4;
        const int l  = pr & 15;

        // A[d] = sum_t Wk[h,d,t] * Q[l, h*32+t];  d0 = lane, d1 = lane+32
        const float* wk0 = Wk + ((size_t)(h * 64 + lane) * 32);
        const float* wk1 = Wk + ((size_t)(h * 64 + lane + 32) * 32);
        const float* q   = smQ + l * NCOL + h * 32;
        float a0 = 0.f, a1 = 0.f;
#pragma unroll 8
        for (int t = 0; t < 32; ++t) {
            const float qt = q[t];
            a0 += wk0[t] * qt;
            a1 += wk1[t] * qt;
        }
        myA[lane]      = a0;
        myA[lane + 32] = a1;
        __syncthreads();

        // QK[c] = (1/sqrt(DK)) * sum_d X[l,c,d] * A[d]   (lanes 0..15 own c)
        {
            const int c = lane & 15;
            const _Float16* xr = smX + l * KDIM + c * 64;
            float qk = 0.f;
#pragma unroll 8
            for (int d = 0; d < 64; ++d)
                qk += (float)xr[d] * myA[d];
            if (lane < 16) myS[lane] = qk * inv_sqrt_dk;
        }
        __syncthreads();

        // softmax over C=16 + attn[d] = sum_c score[c] * X[l,c,d]
        float sc[16];
        float mx = -3.0e38f;
#pragma unroll
        for (int c = 0; c < 16; ++c) { sc[c] = myS[c]; mx = fmaxf(mx, sc[c]); }
        float ssum = 0.f;
#pragma unroll
        for (int c = 0; c < 16; ++c) { sc[c] = __expf(sc[c] - mx); ssum += sc[c]; }
        const float rs = 1.0f / ssum;

        float o0 = 0.f, o1 = 0.f;
#pragma unroll
        for (int c = 0; c < 16; ++c) {
            const _Float16* xr = smX + l * KDIM + c * 64;
            o0 += sc[c] * (float)xr[lane];
            o1 += sc[c] * (float)xr[lane + 32];
        }
        const size_t ob = ((size_t)n * LL + (size_t)ltile * 16 + l) * (NHH * DD) + h * 64;
        out[ob + lane]      = o0 * rs;
        out[ob + lane + 32] = o1 * rs;
        __syncthreads();   // protect smA/smS before next pair
    }
}

extern "C" void kernel_launch(void* const* d_in, const int* in_sizes, int n_in,
                              void* d_out, int out_size, void* d_ws, size_t ws_size,
                              hipStream_t stream) {
    const float* x  = (const float*)d_in[0];   // (64, 8192, 64)
    const float* Wq = (const float*)d_in[1];   // (8, 16, 64, 32)
    const float* Wk = (const float*)d_in[2];   // (8, 64, 32)
    float* out = (float*)d_out;                // (64, 512, 512)
    _Float16* Wt = (_Float16*)d_ws;            // 512 KB packed f16 weights

    wq_pack_f16<<<(NHH * DKK * KDIM + 255) / 256, 256, 0, stream>>>(Wq, Wt);
    attn_fused<<<NB * (LL / 16), 512, 0, stream>>>(x, Wt, Wk, out);
}
